// ComplexAttention_77618648973385
// MI455X (gfx1250) — matrix operand order repaired and checked by reading, not compiled, not run on hardware
//
#include <hip/hip_runtime.h>
#include <hip/hip_bf16.h>
#include <math.h>

// ---------------------------------------------------------------------------
// ComplexAttention == plain attention: scores = (Q.K^T)/sqrt(128),
// W = softmax(scores), out = W @ V ; must also emit W (fp32) to d_out.
// B=4, S=4096, H=256. Outputs: [output (4,4096,256) | weights (4,4096,4096)].
//
// QT=32 query rows per workgroup: 32x4104 bf16 score tile lives in the 320KB
// WGP LDS; every K / V^T B-fragment load feeds TWO wmma ops (A reuse), halving
// L2 traffic vs a 16-row tile. __launch_bounds__(256,1) gives the allocator
// the full VGPR budget (1 WG/WGP anyway due to LDS) so the 16 resident
// A-fragments do NOT spill to scratch.
// ---------------------------------------------------------------------------

typedef __attribute__((ext_vector_type(16))) __bf16 v16bf;
typedef __attribute__((ext_vector_type(8)))  __bf16 v8bf;
typedef __attribute__((ext_vector_type(4)))  __bf16 v4bf;
typedef __attribute__((ext_vector_type(8)))  float  v8f;

#define B_    4
#define S_    4096
#define H_    256
#define QT    32            // query rows per workgroup (two 16-row WMMA tiles)
#define PITCH 4104          // LDS row pitch in bf16 (4096+8 -> conflict-free b128)
#define INV_SCALE 0.08838834764831845f   // 1/sqrt(128)

#define SMEM_W_BYTES   (QT * PITCH * 2)          // 262656 B bf16 score/weight tile
#define SMEM_RED_BYTES (2 * QT * 8 * 4)          // 2x 32x8 f32 reduction scratch
#define SMEM_BYTES     (SMEM_W_BYTES + SMEM_RED_BYTES)

union ABfrag { v16bf v; struct { v8bf lo, hi; } h; };

// --------------------------- fp32 -> bf16 (flat) ---------------------------
__global__ void cvt_bf16_kernel(const float* __restrict__ in,
                                __bf16* __restrict__ out, int n4) {
    int idx = blockIdx.x * blockDim.x + threadIdx.x;
    if (idx < n4) {
        float4 f = reinterpret_cast<const float4*>(in)[idx];
        v4bf o = { (__bf16)f.x, (__bf16)f.y, (__bf16)f.z, (__bf16)f.w };
        reinterpret_cast<v4bf*>(out)[idx] = o;
    }
}

// ----------------------- V (b,s,d) -> V^T bf16 (b,d,s) ---------------------
__global__ void cvt_vt_kernel(const float* __restrict__ V,
                              __bf16* __restrict__ VT) {
    __shared__ float tile[32][33];
    const int b  = blockIdx.z;
    const int s0 = blockIdx.x * 32;
    const int d0 = blockIdx.y * 32;
    for (int r = threadIdx.y; r < 32; r += 8)
        tile[r][threadIdx.x] =
            V[((size_t)b * S_ + s0 + r) * H_ + d0 + threadIdx.x];
    __syncthreads();
    for (int r = threadIdx.y; r < 32; r += 8)
        VT[((size_t)b * H_ + d0 + r) * S_ + s0 + threadIdx.x] =
            (__bf16)tile[threadIdx.x][r];
}

// ------------------------------ main attention -----------------------------
// grid = (S/QT, B), block = 256 (8 wave32).  Dynamic LDS = SMEM_BYTES.
// launch_bounds(256, 1): LDS already restricts to 1 WG/WGP; let the register
// allocator use the whole file instead of spilling A-fragments.
__global__ void __launch_bounds__(256, 1)
attn_kernel(const __bf16* __restrict__ Qb,
            const __bf16* __restrict__ Kb,
            const __bf16* __restrict__ VTb,
            float* __restrict__ outO,
            float* __restrict__ outW) {
    extern __shared__ __align__(16) char smem[];
    __bf16* W    = reinterpret_cast<__bf16*>(smem);
    float*  red  = reinterpret_cast<float*>(smem + SMEM_W_BYTES);
    float*  red2 = red + QT * 8;

    const int tid  = threadIdx.x;
    const int wave = tid >> 5;
    const int lane = tid & 31;
    const int m    = lane & 15;     // row/col within 16x16 tile
    const int half = lane >> 4;     // ISA 16-bit operand half select

    const int b  = blockIdx.y;
    const int qb = blockIdx.x * QT;

    // ---- Phase 1: scores tile (32 x 4096) = Q_tile (32x256) . K^T
    // Two A-fragment sets (q rows m and m+16); each K B-fragment is reused
    // for both wmma ops -> halves L2 read traffic.
    // A layout (ISA 7.12.2, 16-bit A 16x32): lane row = m; elems 0..7 at
    // kd = ks*32 + half*8 .. +7 ; elems 8..15 at kd = ks*32 + 16 + half*8 ..+7
    ABfrag afr0[8], afr1[8];
    {
        const __bf16* Qrow0 = Qb + ((size_t)b * S_ + qb + m) * H_;
        const __bf16* Qrow1 = Qrow0 + 16 * H_;
        #pragma unroll
        for (int ks = 0; ks < 8; ++ks) {
            const __bf16* p0 = Qrow0 + ks * 32 + half * 8;
            const __bf16* p1 = Qrow1 + ks * 32 + half * 8;
            afr0[ks].h.lo = *reinterpret_cast<const v8bf*>(p0);
            afr0[ks].h.hi = *reinterpret_cast<const v8bf*>(p0 + 16);
            afr1[ks].h.lo = *reinterpret_cast<const v8bf*>(p1);
            afr1[ks].h.hi = *reinterpret_cast<const v8bf*>(p1 + 16);
        }
    }

    for (int t = wave; t < S_ / 16; t += 8) {
        // B layout (32x16, 16-bit): lane col n = m (key), elems 0..7 at kd =
        // ks*32 + half*8 ..+7 ; elems 8..15 at kd = ks*32 + 16 + half*8 ..+7
        const int key = t * 16 + m;
        const __bf16* Krow = Kb + ((size_t)b * S_ + key) * H_ + half * 8;
        __builtin_prefetch(Krow + 128 * H_, 0, 1);   // next strided tile
        v8f acc0 = {}, acc1 = {};
        #pragma unroll
        for (int ks = 0; ks < 8; ++ks) {
            ABfrag bf;
            bf.h.lo = *reinterpret_cast<const v8bf*>(Krow + ks * 32);
            bf.h.hi = *reinterpret_cast<const v8bf*>(Krow + ks * 32 + 16);
            acc0 = __builtin_amdgcn_wmma_f32_16x16x32_bf16(
                false, afr0[ks].v, false, bf.v, (short)0, acc0, false, false);
            acc1 = __builtin_amdgcn_wmma_f32_16x16x32_bf16(
                false, afr1[ks].v, false, bf.v, (short)0, acc1, false, false);
        }
        // D layout: elem v -> row half*8+v, col n = m. Store scaled bf16.
        #pragma unroll
        for (int v = 0; v < 8; ++v) {
            W[(half * 8 + v)        * PITCH + t * 16 + m] =
                (__bf16)(acc0[v] * INV_SCALE);
            W[(16 + half * 8 + v)   * PITCH + t * 16 + m] =
                (__bf16)(acc1[v] * INV_SCALE);
        }
    }
    __syncthreads();

    // ---- Phase 2: row softmax in LDS; fp32 weights -> global, bf16 in place
    {
        const int r = tid >> 3;        // row 0..31
        const int i = tid & 7;         // 8 threads cooperate on one row
        __bf16* Wr = W + r * PITCH;

        float mx = -INFINITY;
        for (int j = 0; j < 512; ++j)
            mx = fmaxf(mx, (float)Wr[i + 8 * j]);
        red[r * 8 + i] = mx;
        __syncthreads();
        float rmax = -INFINITY;
        #pragma unroll
        for (int k = 0; k < 8; ++k) rmax = fmaxf(rmax, red[r * 8 + k]);

        float psum = 0.f;
        for (int j = 0; j < 512; ++j)
            psum += __expf((float)Wr[i + 8 * j] - rmax);
        red2[r * 8 + i] = psum;
        __syncthreads();
        float tot = 0.f;
        #pragma unroll
        for (int k = 0; k < 8; ++k) tot += red2[r * 8 + k];
        const float inv = 1.f / tot;

        float* gW = outW + ((size_t)b * S_ + qb + r) * S_;
        for (int j = 0; j < 512; ++j) {
            const int c = i + 8 * j;
            const float w = __expf((float)Wr[c] - rmax) * inv;
            gW[c] = w;                 // fp32 weights output
            Wr[c] = (__bf16)w;         // normalized bf16 for phase 3
        }
    }
    __syncthreads();

    // ---- Phase 3: O (32x256) = W (32x4096) @ V via V^T; each V^T B-fragment
    // feeds two wmma ops (both q sub-tiles) -> halves L2 read traffic.
    for (int nt = wave; nt < H_ / 16; nt += 8) {
        const __bf16* Wm0  = W + m * PITCH + half * 8;          // A, rows 0..15
        const __bf16* Wm1  = Wm0 + 16 * PITCH;                  // A, rows 16..31
        const __bf16* Vrow =
            VTb + ((size_t)b * H_ + nt * 16 + m) * S_ + half * 8;
        __builtin_prefetch(Vrow, 0, 1);
        v8f acc0 = {}, acc1 = {};
        #pragma unroll 4
        for (int ks = 0; ks < S_ / 32; ++ks) {
            ABfrag a0, a1, bb;
            bb.h.lo = *reinterpret_cast<const v8bf*>(Vrow + ks * 32);
            bb.h.hi = *reinterpret_cast<const v8bf*>(Vrow + ks * 32 + 16);
            a0.h.lo = *reinterpret_cast<const v8bf*>(Wm0 + ks * 32);
            a0.h.hi = *reinterpret_cast<const v8bf*>(Wm0 + ks * 32 + 16);
            a1.h.lo = *reinterpret_cast<const v8bf*>(Wm1 + ks * 32);
            a1.h.hi = *reinterpret_cast<const v8bf*>(Wm1 + ks * 32 + 16);
            acc0 = __builtin_amdgcn_wmma_f32_16x16x32_bf16(
                false, a0.v, false, bb.v, (short)0, acc0, false, false);
            acc1 = __builtin_amdgcn_wmma_f32_16x16x32_bf16(
                false, a1.v, false, bb.v, (short)0, acc1, false, false);
        }
        float* gO = outO + ((size_t)b * S_ + qb) * H_ + nt * 16 + m;
        #pragma unroll
        for (int v = 0; v < 8; ++v) {
            gO[(size_t)(half * 8 + v)      * H_] = acc0[v];
            gO[(size_t)(16 + half * 8 + v) * H_] = acc1[v];
        }
    }
}

// ---------------------------------------------------------------------------
extern "C" void kernel_launch(void* const* d_in, const int* in_sizes, int n_in,
                              void* d_out, int out_size, void* d_ws, size_t ws_size,
                              hipStream_t stream) {
    const float* Q = (const float*)d_in[0];
    const float* K = (const float*)d_in[1];
    const float* V = (const float*)d_in[2];

    const size_t NE = (size_t)B_ * S_ * H_;          // 4,194,304 elems
    __bf16* Qb  = (__bf16*)d_ws;
    __bf16* Kb  = Qb + NE;
    __bf16* VTb = Kb + NE;                            // total 24 MiB bf16 in ws

    float* outO = (float*)d_out;
    float* outW = outO + NE;

    // 1) precision/layout staging
    {
        const int n4 = (int)(NE / 4);
        dim3 g((n4 + 255) / 256), blk(256);
        cvt_bf16_kernel<<<g, blk, 0, stream>>>(Q, Qb, n4);
        cvt_bf16_kernel<<<g, blk, 0, stream>>>(K, Kb, n4);
        cvt_vt_kernel<<<dim3(S_ / 32, H_ / 32, B_), dim3(32, 8), 0, stream>>>(V, VTb);
    }

    // 2) fused scores + softmax + PV, one 32-row query tile per workgroup
    static int attr_set = 0;  // idempotent & deterministic; safe across calls
    if (!attr_set) {
        hipFuncSetAttribute(reinterpret_cast<const void*>(attn_kernel),
                            hipFuncAttributeMaxDynamicSharedMemorySize,
                            SMEM_BYTES);
        attr_set = 1;
    }
    attn_kernel<<<dim3(S_ / QT, B_), dim3(256), SMEM_BYTES, stream>>>(
        Qb, Kb, VTb, outO, outW);
}
